// GraphSAGE_5299989643916
// MI455X (gfx1250) — matrix operand order repaired and checked by reading, not compiled, not run on hardware
//
#include <hip/hip_runtime.h>
#include <hip/hip_bf16.h>
#include <stdint.h>

typedef __attribute__((ext_vector_type(16))) _Float16 v16h;
typedef __attribute__((ext_vector_type(8)))  _Float16 v8h;
typedef __attribute__((ext_vector_type(4)))  _Float16 v4h;
typedef __attribute__((ext_vector_type(8)))  float    v8f;
typedef __attribute__((ext_vector_type(4)))  float    v4f;

// ---------------- utility kernels ----------------

__global__ void k_zero_f32(float* __restrict__ p, long n) {
  long i = (long)blockIdx.x * blockDim.x + threadIdx.x;
  if (i < n) p[i] = 0.0f;
}

__global__ void k_cvt_f16(const float* __restrict__ s, _Float16* __restrict__ d, int n) {
  int i = blockIdx.x * blockDim.x + threadIdx.x;
  if (i < n) d[i] = (_Float16)s[i];
}

__global__ void k_count(const int* __restrict__ dst, float* __restrict__ cnt, int E) {
  int e = blockIdx.x * blockDim.x + threadIdx.x;
  if (e < E) atomicAdd(cnt + dst[e], 1.0f);
}

// mean_h[i] = f16(agg[i] / max(cnt[i/128], 1))
__global__ void k_mean_f16(const float* __restrict__ agg,
                           const float* __restrict__ cnt,
                           _Float16* __restrict__ meanh, long n) {
  long i = (long)blockIdx.x * blockDim.x + threadIdx.x;
  if (i < n) {
    float inv = 1.0f / fmaxf(cnt[i >> 7], 1.0f);
    meanh[i] = (_Float16)(agg[i] * inv);
  }
}

// one wave (32 lanes) per edge, 4 channels per lane (128 ch total)
__global__ void k_scatter_f32(const float* __restrict__ x,
                              const int* __restrict__ src,
                              const int* __restrict__ dst,
                              float* __restrict__ agg, int E) {
  long gid = (long)blockIdx.x * blockDim.x + threadIdx.x;
  int e = (int)(gid >> 5);
  if (e >= E) return;
  int c = ((int)gid & 31) << 2;
  long s = src[e], d = dst[e];
  v4f v = *(const v4f*)(x + s * 128 + c);
  float* p = agg + d * 128 + c;
  atomicAdd(p + 0, v[0]);
  atomicAdd(p + 1, v[1]);
  atomicAdd(p + 2, v[2]);
  atomicAdd(p + 3, v[3]);
}

__global__ void k_scatter_f16(const _Float16* __restrict__ h,
                              const int* __restrict__ src,
                              const int* __restrict__ dst,
                              float* __restrict__ agg, int E) {
  long gid = (long)blockIdx.x * blockDim.x + threadIdx.x;
  int e = (int)(gid >> 5);
  if (e >= E) return;
  int c = ((int)gid & 31) << 2;
  long s = src[e], d = dst[e];
  v4h v = *(const v4h*)(h + s * 128 + c);
  float* p = agg + d * 128 + c;
  atomicAdd(p + 0, (float)v[0]);
  atomicAdd(p + 1, (float)v[1]);
  atomicAdd(p + 2, (float)v[2]);
  atomicAdd(p + 3, (float)v[3]);
}

// ---------------- fused SAGE GEMM via WMMA ----------------
// out[m, n] = sum_k mean[m,k]*Wl[n,k] + sum_k root[m,k]*Wr[n,k] + bias[n]
// One wave computes a 16x32 output tile (2 accumulators sharing A fetches).
// A layout (16-bit 16x32, contiguous k): lane L holds row (L&15); elems 0..7 =
// K kk+(L>>4)*16.. ; elems 8..15 = K kk+(L>>4)*16+8.. .
// B layout (32x16): lane L holds col (L&15); elem j = K kk+(L>>4)*16+j.
// C/D: acc[v] -> row tile+v+(L>>4)*8, col = lane&15.
template <int NCOL, bool A2F16, bool RELU, bool OUTF16>
__global__ void k_sage_wmma(const _Float16* __restrict__ meanh,  // [N,128] f16
                            const void* __restrict__ root,       // [N,128] f32 or f16
                            const _Float16* __restrict__ Wl,     // [NCOL,128] f16
                            const _Float16* __restrict__ Wr,     // [NCOL,128] f16
                            const float* __restrict__ bias,      // [NCOL]
                            void* __restrict__ out) {            // [N,NCOL]
  constexpr int K = 128;
  const int lane = threadIdx.x;
  const int r  = lane & 15;
  const int hi = lane >> 4;
  const long rowA = (long)blockIdx.x * 16 + r;
  const int  col0 = blockIdx.y * 32 + r;
  const int  col1 = col0 + 16;

  const _Float16* mrow = meanh + rowA * K;
  const float*    xr32 = (const float*)root + rowA * K;
  const _Float16* xr16 = (const _Float16*)root + rowA * K;
  const _Float16* wl0  = Wl + (long)col0 * K;
  const _Float16* wl1  = Wl + (long)col1 * K;
  const _Float16* wr0  = Wr + (long)col0 * K;
  const _Float16* wr1  = Wr + (long)col1 * K;

  v8f acc0 = {}, acc1 = {};
#pragma unroll
  for (int kk = 0; kk < K; kk += 32) {
    const int kb0 = kk + hi * 16;       // A chunk 1 (16B aligned)
    const int kb1 = kk + hi * 16 + 8;   // A chunk 2
    const int kbB = kk + hi * 16;       // B chunk (32B aligned)

    // A from mean (f16, aligned vector loads)
    v16h a;
    {
      v8h lo = *(const v8h*)(mrow + kb0);
      v8h hi8 = *(const v8h*)(mrow + kb1);
#pragma unroll
      for (int j = 0; j < 8; ++j) { a[j] = lo[j]; a[j + 8] = hi8[j]; }
    }
    v16h b0 = *(const v16h*)(wl0 + kbB);
    v16h b1 = *(const v16h*)(wl1 + kbB);
    acc0 = __builtin_amdgcn_wmma_f32_16x16x32_f16(false, a, false, b0,
                                                  (short)0, acc0, false, false);
    acc1 = __builtin_amdgcn_wmma_f32_16x16x32_f16(false, a, false, b1,
                                                  (short)0, acc1, false, false);

    // A from root (x f32 for layer1, h f16 for layer2)
    v16h a2;
    if (A2F16) {
      v8h lo = *(const v8h*)(xr16 + kb0);
      v8h hi8 = *(const v8h*)(xr16 + kb1);
#pragma unroll
      for (int j = 0; j < 8; ++j) { a2[j] = lo[j]; a2[j + 8] = hi8[j]; }
    } else {
      v4f f0 = *(const v4f*)(xr32 + kb0);
      v4f f1 = *(const v4f*)(xr32 + kb0 + 4);
      v4f f2 = *(const v4f*)(xr32 + kb1);
      v4f f3 = *(const v4f*)(xr32 + kb1 + 4);
#pragma unroll
      for (int j = 0; j < 4; ++j) {
        a2[j]      = (_Float16)f0[j];
        a2[j + 4]  = (_Float16)f1[j];
        a2[j + 8]  = (_Float16)f2[j];
        a2[j + 12] = (_Float16)f3[j];
      }
    }
    v16h b2 = *(const v16h*)(wr0 + kbB);
    v16h b3 = *(const v16h*)(wr1 + kbB);
    acc0 = __builtin_amdgcn_wmma_f32_16x16x32_f16(false, a2, false, b2,
                                                  (short)0, acc0, false, false);
    acc1 = __builtin_amdgcn_wmma_f32_16x16x32_f16(false, a2, false, b3,
                                                  (short)0, acc1, false, false);
  }

  const float bv0 = bias[col0];
  const float bv1 = bias[col1];
#pragma unroll
  for (int v = 0; v < 8; ++v) {
    const long m = (long)blockIdx.x * 16 + v + hi * 8;
    float val0 = acc0[v] + bv0;
    float val1 = acc1[v] + bv1;
    if (RELU) { val0 = fmaxf(val0, 0.0f); val1 = fmaxf(val1, 0.0f); }
    if (OUTF16) {
      ((_Float16*)out)[m * NCOL + col0] = (_Float16)val0;
      ((_Float16*)out)[m * NCOL + col1] = (_Float16)val1;
    } else {
      ((float*)out)[m * NCOL + col0] = val0;
      ((float*)out)[m * NCOL + col1] = val1;
    }
  }
}

// ---------------- launch ----------------

extern "C" void kernel_launch(void* const* d_in, const int* in_sizes, int n_in,
                              void* d_out, int out_size, void* d_ws, size_t ws_size,
                              hipStream_t stream) {
  const float* x   = (const float*)d_in[0];
  const int*   ei  = (const int*)d_in[1];
  const float* Wl1 = (const float*)d_in[2];
  const float* bl1 = (const float*)d_in[3];
  const float* Wr1 = (const float*)d_in[4];
  const float* Wl2 = (const float*)d_in[5];
  const float* bl2 = (const float*)d_in[6];
  const float* Wr2 = (const float*)d_in[7];

  const int N = in_sizes[0] / 128;      // 100000 (multiple of 16)
  const int E = in_sizes[1] / 2;        // 1600000
  const int* src = ei;
  const int* dst = ei + E;

  char* w = (char*)d_ws;
  float* agg = (float*)w;            w += (size_t)N * 128 * sizeof(float);
  float* cnt = (float*)w;            w += (size_t)N * sizeof(float);
  _Float16* h     = (_Float16*)w;    w += (size_t)N * 128 * sizeof(_Float16);
  _Float16* meanh = (_Float16*)w;    w += (size_t)N * 128 * sizeof(_Float16);
  _Float16* Wl1h = (_Float16*)w;     w += (size_t)128 * 128 * sizeof(_Float16);
  _Float16* Wr1h = (_Float16*)w;     w += (size_t)128 * 128 * sizeof(_Float16);
  _Float16* Wl2h = (_Float16*)w;     w += (size_t)64 * 128 * sizeof(_Float16);
  _Float16* Wr2h = (_Float16*)w;     w += (size_t)64 * 128 * sizeof(_Float16);

  const long nfeat = (long)N * 128;

  // zero agg + cnt (contiguous region)
  long nz = (long)N * 129;
  k_zero_f32<<<(int)((nz + 255) / 256), 256, 0, stream>>>(agg, nz);

  // weights -> f16
  k_cvt_f16<<<(128 * 128 + 255) / 256, 256, 0, stream>>>(Wl1, Wl1h, 128 * 128);
  k_cvt_f16<<<(128 * 128 + 255) / 256, 256, 0, stream>>>(Wr1, Wr1h, 128 * 128);
  k_cvt_f16<<<(64 * 128 + 255) / 256, 256, 0, stream>>>(Wl2, Wl2h, 64 * 128);
  k_cvt_f16<<<(64 * 128 + 255) / 256, 256, 0, stream>>>(Wr2, Wr2h, 64 * 128);

  // in-degree counts
  k_count<<<(E + 255) / 256, 256, 0, stream>>>(dst, cnt, E);

  // layer 1 aggregation (gather x -> atomic add into agg)
  long tsc = (long)E * 32;
  k_scatter_f32<<<(int)((tsc + 255) / 256), 256, 0, stream>>>(x, src, dst, agg, E);

  // mean -> f16 (read once, so GEMM A-loads are pure f16)
  k_mean_f16<<<(int)((nfeat + 255) / 256), 256, 0, stream>>>(agg, cnt, meanh, nfeat);

  // layer 1 fused GEMM + bias + ReLU -> h (f16); 16x32 tiles per wave
  dim3 g1(N / 16, 128 / 32);
  k_sage_wmma<128, false, true, true><<<g1, 32, 0, stream>>>(
      meanh, (const void*)x, Wl1h, Wr1h, bl1, (void*)h);

  // re-zero agg for layer 2
  k_zero_f32<<<(int)((nfeat + 255) / 256), 256, 0, stream>>>(agg, nfeat);

  // layer 2 aggregation (gather h f16 -> atomic add into agg)
  k_scatter_f16<<<(int)((tsc + 255) / 256), 256, 0, stream>>>(h, src, dst, agg, E);

  // mean -> f16 for layer 2
  k_mean_f16<<<(int)((nfeat + 255) / 256), 256, 0, stream>>>(agg, cnt, meanh, nfeat);

  // layer 2 fused GEMM + bias -> out (f32, N x 64); 16x32 tiles per wave
  dim3 g2(N / 16, 64 / 32);
  k_sage_wmma<64, true, false, false><<<g2, 32, 0, stream>>>(
      meanh, (const void*)h, Wl2h, Wr2h, bl2, d_out);
}
